// MultiHeadAttention_41875931136711
// MI455X (gfx1250) — matrix operand order repaired and checked by reading, not compiled, not run on hardware
//
#include <hip/hip_runtime.h>
#include <hip/hip_bf16.h>

// ---------------------------------------------------------------------------
// MultiHeadAttention for MI455X (gfx1250): bf16 WMMA + TDM(LDS) pipeline.
//   B=4, T=2048, C=1024, H=16, D=64
// ---------------------------------------------------------------------------

#define BB 4
#define TT 2048
#define CC 1024
#define HH 16
#define DD 64
#define MM (BB * TT)          // 8192 rows
#define L2E 1.44269504088896340736f

typedef __attribute__((ext_vector_type(16))) __bf16 v16bf;
typedef __attribute__((ext_vector_type(8)))  float  v8f;
typedef __attribute__((ext_vector_type(8)))  unsigned int v8u;
typedef __attribute__((ext_vector_type(4)))  unsigned int v4u;
typedef __attribute__((ext_vector_type(8)))  int v8i;
typedef __attribute__((ext_vector_type(4)))  int v4i;

__device__ __forceinline__ unsigned short f2bf(float f) {
    unsigned int u = __float_as_uint(f);
    u += 0x7FFFu + ((u >> 16) & 1u);          // round-to-nearest-even
    return (unsigned short)(u >> 16);
}

__device__ __forceinline__ v16bf bcast_bf(v8u u) {
    return __builtin_bit_cast(v16bf, u);
}

__device__ __forceinline__ v8f wmma_bf16(v16bf a, v16bf b, v8f c) {
    return __builtin_amdgcn_wmma_f32_16x16x32_bf16(
        /*neg_a=*/false, a, /*neg_b=*/false, b,
        /*c_mod=*/(short)0, c, /*reuse_a=*/false, /*reuse_b=*/false);
}

// ---------------------------------------------------------------------------
// TDM: 2-D bf16 tile (tile_d0 contiguous elems x tile_d1 rows) -> LDS.
// D# per ISA 8.3/8.4: group0 {count=1, lds_addr, global_addr, type=2},
// group1 {data_size=1(2B), tensor dims, tile dims, dim0 stride}. Groups 2/3
// and the trailing group are zero (tile_dim2=0 -> 2-D tensor).
// This toolchain exposes the 6-arg builtin: (g0, g1, g2, g3, g4, cpol).
// ---------------------------------------------------------------------------
__device__ __forceinline__ void tdm_load_2d(void* lds_ptr, const void* gptr,
                                            unsigned int tile_d0, unsigned int tile_d1,
                                            unsigned int tensor_d0, unsigned int tensor_d1,
                                            unsigned long long row_stride_elems) {
    unsigned long long ga = (unsigned long long)gptr;
    unsigned int la = (unsigned int)(unsigned long long)lds_ptr;  // low 32 = LDS offset
    v4u g0;
    g0[0] = 1u;                                                    // count=1
    g0[1] = la;                                                    // lds_addr
    g0[2] = (unsigned int)ga;                                      // global_addr lo
    g0[3] = (unsigned int)((ga >> 32) & 0x01FFFFFFu) | (2u << 30); // addr hi | type=2
    v8i g1;
    g1[0] = (int)(1u << 16);                                       // data_size=1 (2B)
    g1[1] = (int)((tensor_d0 & 0xFFFFu) << 16);                    // tensor_dim0[15:0]
    g1[2] = (int)((tensor_d0 >> 16) | ((tensor_d1 & 0xFFFFu) << 16));
    g1[3] = (int)((tensor_d1 >> 16) | (tile_d0 << 16));            // tile_dim0
    g1[4] = (int)(tile_d1 & 0xFFFFu);                              // tile_dim1, tile_dim2=0
    g1[5] = (int)(unsigned int)(row_stride_elems & 0xFFFFFFFFull); // dim0_stride lo
    g1[6] = (int)(unsigned int)((row_stride_elems >> 32) & 0xFFFFull);
    g1[7] = 0;
    v4i z4 = {0, 0, 0, 0};
    v8i z8 = {0, 0, 0, 0, 0, 0, 0, 0};
    __builtin_amdgcn_tensor_load_to_lds(g0, g1, z4, z4, z8, 0);
}

// A-matrix 16x32 bf16 fragment (ISA 7.12.2): lane = M, lane>=16 selects K-half;
// per-VGPR k offsets {0,2,4,6,16,18,20,22} + 8*khalf (pairs contiguous).
__device__ __forceinline__ v8u ld_fragA(const unsigned short* p, int rstride, int lane) {
    int r = lane & 15, kh = lane >> 4;
    const unsigned short* q = p + (size_t)r * rstride + kh * 8;
    v8u o;
    o[0] = *(const unsigned int*)(q + 0);
    o[1] = *(const unsigned int*)(q + 2);
    o[2] = *(const unsigned int*)(q + 4);
    o[3] = *(const unsigned int*)(q + 6);
    o[4] = *(const unsigned int*)(q + 16);
    o[5] = *(const unsigned int*)(q + 18);
    o[6] = *(const unsigned int*)(q + 20);
    o[7] = *(const unsigned int*)(q + 22);
    return o;
}

// B-matrix 32x16 bf16 fragment: lane = N column; same k order as A:
// per-VGPR k offsets {0,2,4,6,16,18,20,22} + 8*khalf (packed pairs per VGPR).
__device__ __forceinline__ v8u ld_fragB(const unsigned short* p, int rstride, int lane) {
    int r = lane & 15, kh = lane >> 4;
    const unsigned short* q = p + (size_t)r * rstride + kh * 8;
    v8u o;
#pragma unroll
    for (int i = 0; i < 8; ++i)
        o[i] = *(const unsigned int*)(q + 2 * i + (i >= 4 ? 8 : 0));
    return o;
}

// ---------------------------------------------------------------------------
// fp32 -> bf16 convert
// ---------------------------------------------------------------------------
__global__ __launch_bounds__(256) void k_f32_to_bf16(const float* __restrict__ in,
                                                     unsigned short* __restrict__ out,
                                                     int n) {
    for (int i = blockIdx.x * blockDim.x + threadIdx.x; i < n;
         i += gridDim.x * blockDim.x)
        out[i] = f2bf(in[i]);
}

// ---------------------------------------------------------------------------
// WMMA GEMM:  out[M,N] = A[M,K](bf16) @ W[N,K]^T(bf16) + bias, M=8192 N=K=1024
// Block = 4 waves = 128x64 tile; wave = 32x64 (8 WMMAs / k-step).
// TDM double-buffers the A (128x32) and B (64x32) tiles into LDS.
// mode 0: bf16 out in [B,H,T,D]   (Q with scale=1/8, K)
// mode 1: bf16 out in [B,H,D,T]   (V transposed for attention B-fragments)
// mode 2: fp32 out row-major [M,N] (final projection)
// ---------------------------------------------------------------------------
__global__ __launch_bounds__(128) void k_gemm_wmma(const unsigned short* __restrict__ A,
                                                   const unsigned short* __restrict__ W,
                                                   const float* __restrict__ bias,
                                                   void* __restrict__ out,
                                                   int mode, float scale) {
    __shared__ unsigned short Abuf[2][128 * 32];   // 8 KB x2
    __shared__ unsigned short Bbuf[2][64 * 32];    // 4 KB x2

    const int lane = threadIdx.x & 31;
    const int wave = threadIdx.x >> 5;
    const int TILES_N = CC / 64;                   // 16
    const int m0 = (blockIdx.x / TILES_N) * 128;
    const int n0 = (blockIdx.x % TILES_N) * 64;
    const int NIT = CC / 32;                       // 32 k-steps

    v8f acc[2][4];
#pragma unroll
    for (int s = 0; s < 2; ++s)
#pragma unroll
        for (int t = 0; t < 4; ++t)
#pragma unroll
            for (int i = 0; i < 8; ++i) acc[s][t][i] = 0.0f;

    if (wave == 0) {   // prologue: stage k-step 0
        tdm_load_2d(&Abuf[0][0], A + (size_t)m0 * CC, 32, 128, CC, MM, CC);
        tdm_load_2d(&Bbuf[0][0], W + (size_t)n0 * CC, 32, 64,  CC, CC, CC);
    }

    for (int j = 0; j < NIT; ++j) {
        const int cur = j & 1;
        if (wave == 0) {
            if (j + 1 < NIT) {
                const int kb = (j + 1) * 32;
                tdm_load_2d(&Abuf[cur ^ 1][0], A + (size_t)m0 * CC + kb, 32, 128, CC, MM, CC);
                tdm_load_2d(&Bbuf[cur ^ 1][0], W + (size_t)n0 * CC + kb, 32, 64,  CC, CC, CC);
                __builtin_amdgcn_s_wait_tensorcnt(2);
            } else {
                __builtin_amdgcn_s_wait_tensorcnt(0);
            }
        }
        __syncthreads();                           // buf[cur] ready for all waves

        v16bf av0 = bcast_bf(ld_fragA(&Abuf[cur][(wave * 32 + 0)  * 32], 32, lane));
        v16bf av1 = bcast_bf(ld_fragA(&Abuf[cur][(wave * 32 + 16) * 32], 32, lane));
#pragma unroll
        for (int nt = 0; nt < 4; ++nt) {
            v16bf bv = bcast_bf(ld_fragB(&Bbuf[cur][(nt * 16) * 32], 32, lane));
            acc[0][nt] = wmma_bf16(av0, bv, acc[0][nt]);
            acc[1][nt] = wmma_bf16(av1, bv, acc[1][nt]);
        }
        __syncthreads();                           // done with buf[cur] before overwrite
    }

    const int ncol = lane & 15;
    const int rb = (lane >> 4) * 8;
#pragma unroll
    for (int s = 0; s < 2; ++s) {
#pragma unroll
        for (int nt = 0; nt < 4; ++nt) {
            const int ng = n0 + nt * 16 + ncol;
            const float bi = bias[ng];
#pragma unroll
            for (int i = 0; i < 8; ++i) {
                const int r = m0 + wave * 32 + s * 16 + rb + i;   // row in [0,8192)
                const float v = (acc[s][nt][i] + bi) * scale;
                if (mode == 2) {
                    ((float*)out)[(size_t)r * CC + ng] = v;
                } else {
                    const int b = r >> 11, t = r & (TT - 1);
                    const int h = ng >> 6, d = ng & (DD - 1);
                    size_t idx;
                    if (mode == 0)      // [B,H,T,D]
                        idx = (((size_t)(b * HH + h) * TT + t) * DD + d);
                    else                // [B,H,D,T]
                        idx = (((size_t)(b * HH + h) * DD + d) * TT + t);
                    ((unsigned short*)out)[idx] = f2bf(v);
                }
            }
        }
    }
}

// ---------------------------------------------------------------------------
// Flash attention: block = 4 waves = 64 query rows of one (b,h); all 4 waves
// share the K/V stream, which is TDM double-buffered into LDS in 32-key
// blocks. Online softmax per wave; Q pre-scaled by 1/sqrt(D). Emits bf16
// attn-out in [B*T, C] row-major for the final projection GEMM.
// ---------------------------------------------------------------------------
__global__ __launch_bounds__(128) void k_flash_attn(const unsigned short* __restrict__ Qb,
                                                    const unsigned short* __restrict__ Kb,
                                                    const unsigned short* __restrict__ VTb,
                                                    unsigned short* __restrict__ AOb) {
    __shared__ unsigned short kbuf[2][32 * 64];      // 32 keys x 64 d   (4 KB x2)
    __shared__ unsigned short vbuf[2][64 * 32];      // 64 d    x 32 t   (4 KB x2)
    __shared__ unsigned short plds[4][16 * 34];      // per-wave P tile

    const int lane = threadIdx.x & 31;
    const int wave = threadIdx.x >> 5;
    const int gw = blockIdx.x * 4 + wave;            // 0..8191
    const int qt = gw & 127;                         // query tile within head
    const int bh = gw >> 7;                          // b*H + h (uniform per block)
    const int q0 = qt * 16;

    const unsigned short* Qp = Qb + ((size_t)bh * TT + q0) * DD;
    const v16bf qa0 = bcast_bf(ld_fragA(Qp + 0,  DD, lane));
    const v16bf qa1 = bcast_bf(ld_fragA(Qp + 32, DD, lane));

    const unsigned short* Kbase = Kb  + (size_t)bh * TT * DD;  // [T,D]
    const unsigned short* Vbase = VTb + (size_t)bh * DD * TT;  // [D,T]

    v8f o[4];
    float mr[8], lr[8];
#pragma unroll
    for (int t = 0; t < 4; ++t)
#pragma unroll
        for (int i = 0; i < 8; ++i) o[t][i] = 0.0f;
#pragma unroll
    for (int i = 0; i < 8; ++i) { mr[i] = -INFINITY; lr[i] = 0.0f; }

    unsigned short* pl = plds[wave];
    const int ncol = lane & 15;
    const int rb = (lane >> 4) * 8;
    const int NIT = TT / 32;                         // 64 key blocks

    if (wave == 0) {   // prologue: stage key-block 0
        tdm_load_2d(&kbuf[0][0], Kbase, 64, 32, DD, TT, DD);   // 32 keys x 64 d
        tdm_load_2d(&vbuf[0][0], Vbase, 32, 64, TT, DD, TT);   // 64 d x 32 t
    }

    for (int j = 0; j < NIT; ++j) {
        const int cur = j & 1;
        const int kb = j * 32;
        if (wave == 0) {
            if (j + 1 < NIT) {
                tdm_load_2d(&kbuf[cur ^ 1][0], Kbase + (size_t)(kb + 32) * DD, 64, 32, DD, TT, DD);
                tdm_load_2d(&vbuf[cur ^ 1][0], Vbase + (size_t)(kb + 32),      32, 64, TT, DD, TT);
                __builtin_amdgcn_s_wait_tensorcnt(2);
            } else {
                __builtin_amdgcn_s_wait_tensorcnt(0);
            }
        }
        __syncthreads();                             // kbuf/vbuf[cur] ready

        // ---- S = Q @ K^T for 16x32 keys (two n-tiles, chained over d) ----
        v8f z;
#pragma unroll
        for (int i = 0; i < 8; ++i) z[i] = 0.0f;

        v8f s0 = wmma_bf16(qa0, bcast_bf(ld_fragB(&kbuf[cur][0 * 64 + 0],   64, lane)), z);
        s0     = wmma_bf16(qa1, bcast_bf(ld_fragB(&kbuf[cur][0 * 64 + 32],  64, lane)), s0);
        v8f s1 = wmma_bf16(qa0, bcast_bf(ld_fragB(&kbuf[cur][16 * 64 + 0],  64, lane)), z);
        s1     = wmma_bf16(qa1, bcast_bf(ld_fragB(&kbuf[cur][16 * 64 + 32], 64, lane)), s1);

        // ---- online softmax row statistics ----
        float mx[8];
#pragma unroll
        for (int i = 0; i < 8; ++i) mx[i] = fmaxf(s0[i], s1[i]);
#pragma unroll
        for (int off = 1; off <= 8; off <<= 1)
#pragma unroll
            for (int i = 0; i < 8; ++i)
                mx[i] = fmaxf(mx[i], __shfl_xor(mx[i], off, 32));

        float mn[8], al[8];
#pragma unroll
        for (int i = 0; i < 8; ++i) {
            mn[i] = fmaxf(mr[i], mx[i]);
            al[i] = exp2f((mr[i] - mn[i]) * L2E);
            mr[i] = mn[i];
        }
#pragma unroll
        for (int i = 0; i < 8; ++i) {
            s0[i] = exp2f((s0[i] - mn[i]) * L2E);
            s1[i] = exp2f((s1[i] - mn[i]) * L2E);
        }
        float rs[8];
#pragma unroll
        for (int i = 0; i < 8; ++i) rs[i] = s0[i] + s1[i];
#pragma unroll
        for (int off = 1; off <= 8; off <<= 1)
#pragma unroll
            for (int i = 0; i < 8; ++i)
                rs[i] += __shfl_xor(rs[i], off, 32);
#pragma unroll
        for (int i = 0; i < 8; ++i) lr[i] = lr[i] * al[i] + rs[i];
#pragma unroll
        for (int t = 0; t < 4; ++t)
#pragma unroll
            for (int i = 0; i < 8; ++i) o[t][i] *= al[i];

        // ---- re-fragment P: C-layout f32 -> LDS -> A-layout bf16 ----
        // (wave-private region; DS ops from one wave are processed in order)
#pragma unroll
        for (int i = 0; i < 8; ++i) {
            pl[(rb + i) * 34 + ncol]      = f2bf(s0[i]);
            pl[(rb + i) * 34 + 16 + ncol] = f2bf(s1[i]);
        }
        v8u pu;
        {
            const int r = lane & 15, kh = lane >> 4;
            const unsigned short* q = pl + r * 34 + kh * 8;
            pu[0] = *(const unsigned int*)(q + 0);
            pu[1] = *(const unsigned int*)(q + 2);
            pu[2] = *(const unsigned int*)(q + 4);
            pu[3] = *(const unsigned int*)(q + 6);
            pu[4] = *(const unsigned int*)(q + 16);
            pu[5] = *(const unsigned int*)(q + 18);
            pu[6] = *(const unsigned int*)(q + 20);
            pu[7] = *(const unsigned int*)(q + 22);
        }
        const v16bf pa = bcast_bf(pu);

        // ---- O += P @ V  (V tile [d,t] in LDS: B-fragment pairs contiguous) ----
#pragma unroll
        for (int nt = 0; nt < 4; ++nt) {
            v16bf vv = bcast_bf(ld_fragB(&vbuf[cur][(nt * 16) * 32], 32, lane));
            o[nt] = wmma_bf16(pa, vv, o[nt]);
        }
        __syncthreads();                             // done with buf[cur]
    }

    // ---- normalize and emit bf16 [B*T, C] for the output projection ----
    const int b = bh >> 4, h = bh & 15;
#pragma unroll
    for (int i = 0; i < 8; ++i) {
        const float inv = 1.0f / lr[i];
        const int t = q0 + rb + i;
#pragma unroll
        for (int nt = 0; nt < 4; ++nt) {
            const int col = h * DD + nt * 16 + ncol;
            AOb[(size_t)(b * TT + t) * CC + col] = f2bf(o[nt][i] * inv);
        }
    }
}

// ---------------------------------------------------------------------------
// Host-side launch
// ---------------------------------------------------------------------------
extern "C" void kernel_launch(void* const* d_in, const int* in_sizes, int n_in,
                              void* d_out, int out_size, void* d_ws, size_t ws_size,
                              hipStream_t stream) {
    const float* x  = (const float*)d_in[0];
    const float* Wq = (const float*)d_in[1];
    const float* bq = (const float*)d_in[2];
    const float* Wk = (const float*)d_in[3];
    const float* bk = (const float*)d_in[4];
    const float* Wv = (const float*)d_in[5];
    const float* bv = (const float*)d_in[6];
    const float* Wo = (const float*)d_in[7];
    const float* bo = (const float*)d_in[8];

    // workspace carve-up (bf16 = ushort elements)
    unsigned short* ws = (unsigned short*)d_ws;
    unsigned short* Xbf = ws;                 ws += (size_t)MM * CC;   // 16 MB
    unsigned short* Wqb = ws;                 ws += (size_t)CC * CC;   //  2 MB
    unsigned short* Wkb = ws;                 ws += (size_t)CC * CC;
    unsigned short* Wvb = ws;                 ws += (size_t)CC * CC;
    unsigned short* Wob = ws;                 ws += (size_t)CC * CC;
    unsigned short* Qb  = ws;                 ws += (size_t)BB * HH * TT * DD;
    unsigned short* Kb  = ws;                 ws += (size_t)BB * HH * TT * DD;
    unsigned short* VTb = ws;                 ws += (size_t)BB * HH * DD * TT;
    unsigned short* AOb = ws;                 ws += (size_t)MM * CC;

    // 1) downconvert inputs to bf16
    k_f32_to_bf16<<<4096, 256, 0, stream>>>(x,  Xbf, MM * CC);
    k_f32_to_bf16<<<1024, 256, 0, stream>>>(Wq, Wqb, CC * CC);
    k_f32_to_bf16<<<1024, 256, 0, stream>>>(Wk, Wkb, CC * CC);
    k_f32_to_bf16<<<1024, 256, 0, stream>>>(Wv, Wvb, CC * CC);
    k_f32_to_bf16<<<1024, 256, 0, stream>>>(Wo, Wob, CC * CC);

    // 2) QKV projections: (8192/128)*(1024/64) = 1024 blocks of 4 waves
    const int gemm_blocks = (MM / 128) * (CC / 64);
    k_gemm_wmma<<<gemm_blocks, 128, 0, stream>>>(Xbf, Wqb, bq, Qb, 0, 0.125f); // Q/sqrt(D)
    k_gemm_wmma<<<gemm_blocks, 128, 0, stream>>>(Xbf, Wkb, bk, Kb, 0, 1.0f);
    k_gemm_wmma<<<gemm_blocks, 128, 0, stream>>>(Xbf, Wvb, bv, VTb, 1, 1.0f);  // V^T
    // 3) flash attention: B*H*(T/16) = 8192 waves -> 2048 blocks
    k_flash_attn<<<2048, 128, 0, stream>>>(Qb, Kb, VTb, AOb);

    // 4) output projection -> fp32 d_out
    k_gemm_wmma<<<gemm_blocks, 128, 0, stream>>>(AOb, Wob, bo, d_out, 2, 1.0f);
}
